// S4DKernel_simple_63513976374076
// MI455X (gfx1250) — matrix operand order repaired and checked by reading, not compiled, not run on hardware
//
#include <hip/hip_runtime.h>
#include <hip/hip_bf16.h>

typedef __attribute__((ext_vector_type(16))) _Float16 v16h;
typedef __attribute__((ext_vector_type(8)))  float    v8f;

#define H_DIM  512
#define N2_DIM 32
#define L_DIM  4096
#define TILES_PER_WAVE 4
#define A_SCALE     4096.0f
#define INV_A_SCALE (1.0f / 4096.0f)

// One wave computes FOUR consecutive 16x16 output tiles of K[h,:] using
//   l = l1 + 16*l2   (l1 = 0..15 rows, l2 = cols):
//   A[l1, 2n]   =  2*Re(Ceff_n * w_n^l1),  A[l1, 2n+1] = -2*Im(...)
//   B[2n, l2]   =    Re(w_n^(16*l2)),      B[2n+1, l2] =    Im(...)
// A depends only on h -> built once per wave (amortized over 4 tiles).
// B advances between tiles by the recurrence B *= w^256 (f32 complex mul,
// no transcendentals, no cross-lane ops). Contraction K=64 -> two
// v_wmma_f32_16x16x32_f16 per tile, 8 per wave.
__global__ __launch_bounds__(256) void s4d_wmma_kernel(
    const float* __restrict__ C,
    const float* __restrict__ log_dt,
    const float* __restrict__ log_A_real,
    const float* __restrict__ A_imag,
    float* __restrict__ out)
{
    const int lane  = threadIdx.x & 31;
    const int wave  = threadIdx.x >> 5;
    const int wid   = blockIdx.x * 8 + wave;          // 0..2047
    const int h     = wid >> 2;                       // 4 wave-groups per h
    const int t0    = (wid & 3) * TILES_PER_WAVE;     // first l2-tile (of 16)

    const float dt     = __expf(log_dt[h]);
    const bool  hiHalf = (lane >= 16);
    const int   lane16 = lane & 15;

    // ---- Per-lane pole parameters for n = lane (wave covers all 32 n).
    float dAr_l, dAi_l, cr_l, ci_l;
    {
        const int idx = h * N2_DIM + lane;
        const float Ar = -__expf(log_A_real[idx]);
        const float Ai = A_imag[idx];
        dAr_l = Ar * dt;
        dAi_l = Ai * dt;
        const float er = __expf(dAr_l);
        float s, c;
        __sincosf(dAi_l, &s, &c);
        const float Er = er * c - 1.0f;
        const float Ei = er * s;
        const float inv = __builtin_amdgcn_rcpf(Ar * Ar + Ai * Ai);
        const float qr = (Er * Ar + Ei * Ai) * inv;
        const float qi = (Ei * Ar - Er * Ai) * inv;
        const float Ccr = C[2 * idx + 0];
        const float Cci = C[2 * idx + 1];
        cr_l = Ccr * qr - Cci * qi;
        ci_l = Ccr * qi + Cci * qr;
    }

    // ---- A fragments (16x32 f16 each K-half), built ONCE, reused for all
    // tiles. Row M = lane16, power l1 = M.
    // Layout: lane<16 covers K 0..15; lane>=16 covers K 16..31.
    v16h afrag[2];
    {
        const float mPow = (float)lane16;
        #pragma unroll
        for (int kb = 0; kb < 2; ++kb) {
            #pragma unroll
            for (int p = 0; p < 8; ++p) {
                const int n = kb * 16 + (hiHalf ? 8 : 0) + p;
                const float dAr = __shfl(dAr_l, n, 32);
                const float dAi = __shfl(dAi_l, n, 32);
                const float crn = __shfl(cr_l, n, 32);
                const float cin = __shfl(ci_l, n, 32);
                const float er = __expf(dAr * mPow);
                float s, co;
                __sincosf(dAi * mPow, &s, &co);
                const float wr = er * co, wi = er * s;    // w_n^l1
                afrag[kb][2 * p]     = (_Float16)(( 2.0f * A_SCALE) * (crn * wr - cin * wi));
                afrag[kb][2 * p + 1] = (_Float16)((-2.0f * A_SCALE) * (crn * wi + cin * wr));
            }
        }
    }

    // ---- Initial B state (f32) for tile t0, plus per-element step rho=w^256.
    // B layout: lanes 0-15 hold K 0..15 (n = nb..nb+7), lanes 16-31 K 16..31.
    float Br[2][8], Bi[2][8], Rr[2][8], Ri[2][8];
    {
        const float pw = (float)(256 * t0 + 16 * lane16);  // 16 * col0
        #pragma unroll
        for (int kb = 0; kb < 2; ++kb) {
            #pragma unroll
            for (int p = 0; p < 8; ++p) {
                const int n = kb * 16 + (hiHalf ? 8 : 0) + p;
                const float dAr = __shfl(dAr_l, n, 32);
                const float dAi = __shfl(dAi_l, n, 32);
                const float er = __expf(dAr * pw);
                float s, co;
                __sincosf(dAi * pw, &s, &co);
                Br[kb][p] = er * co;                       // Re(w_n^pw)
                Bi[kb][p] = er * s;                        // Im(w_n^pw)
                const float er2 = __expf(dAr * 256.0f);
                float s2, c2;
                __sincosf(dAi * 256.0f, &s2, &c2);
                Rr[kb][p] = er2 * c2;                      // Re(w_n^256)
                Ri[kb][p] = er2 * s2;                      // Im(w_n^256)
            }
        }
    }

    const int rowBase = hiHalf ? 8 : 0;

    #pragma unroll
    for (int k = 0; k < TILES_PER_WAVE; ++k) {
        // Pack B fragments (f16) from running f32 state; two WMMAs (K=64).
        v8f acc = {};
        #pragma unroll
        for (int kb = 0; kb < 2; ++kb) {
            v16h b;
            #pragma unroll
            for (int p = 0; p < 8; ++p) {
                b[2 * p]     = (_Float16)Br[kb][p];
                b[2 * p + 1] = (_Float16)Bi[kb][p];
            }
            acc = __builtin_amdgcn_wmma_f32_16x16x32_f16(
                /*neg_a=*/false, afrag[kb], /*neg_b=*/false, b,
                /*c_mod=*/(short)0, acc, /*reuse_a=*/false, /*reuse_b=*/false);
        }

        // Store: C/D layout VGPR r -> M = r (+8 upper lanes), N = lane16.
        // l = M + 16*col -> lane's 8 results contiguous: two b128 stores.
        const int col = (t0 + k) * 16 + lane16;
        float* o = out + (size_t)h * L_DIM + col * 16 + rowBase;
        float4 v0, v1;
        v0.x = acc[0] * INV_A_SCALE; v0.y = acc[1] * INV_A_SCALE;
        v0.z = acc[2] * INV_A_SCALE; v0.w = acc[3] * INV_A_SCALE;
        v1.x = acc[4] * INV_A_SCALE; v1.y = acc[5] * INV_A_SCALE;
        v1.z = acc[6] * INV_A_SCALE; v1.w = acc[7] * INV_A_SCALE;
        *(float4*)(o)     = v0;
        *(float4*)(o + 4) = v1;

        // Advance B to next tile: B *= rho  (complex, f32, VALU-only).
        if (k != TILES_PER_WAVE - 1) {
            #pragma unroll
            for (int kb = 0; kb < 2; ++kb) {
                #pragma unroll
                for (int p = 0; p < 8; ++p) {
                    const float br = Br[kb][p], bi = Bi[kb][p];
                    Br[kb][p] = br * Rr[kb][p] - bi * Ri[kb][p];
                    Bi[kb][p] = br * Ri[kb][p] + bi * Rr[kb][p];
                }
            }
        }
    }
}

extern "C" void kernel_launch(void* const* d_in, const int* in_sizes, int n_in,
                              void* d_out, int out_size, void* d_ws, size_t ws_size,
                              hipStream_t stream) {
    // Input order per setup_inputs(): L (scalar), C (H,N2,2), log_dt (H),
    // log_A_real (H,N2), A_imag (H,N2). Output: K (H,L) f32.
    const float* C          = (const float*)d_in[1];
    const float* log_dt     = (const float*)d_in[2];
    const float* log_A_real = (const float*)d_in[3];
    const float* A_imag     = (const float*)d_in[4];
    float* out = (float*)d_out;

    // 512 h * 4 wave-groups = 2048 waves; 8 waves (256 threads) per block.
    const int blocks = (H_DIM * 4) / 8;   // 256
    s4d_wmma_kernel<<<blocks, 256, 0, stream>>>(C, log_dt, log_A_real, A_imag, out);
}
